// GAMMANet_49392123904380
// MI455X (gfx1250) — matrix operand order repaired and checked by reading, not compiled, not run on hardware
//
#include <hip/hip_runtime.h>
#include <hip/hip_bf16.h>
#include <math.h>

// ---- problem constants (from reference) ----
#define Bb   4
#define Tt   12
#define Nn   207
#define Dd   152
#define Hh   4
#define Cc   152
#define Ee   1722
#define DINc 304
#define Sc   16
#define Rc   10
#define SPDc 288
#define ETOT (Ee + Nn)        // edges + self loops = 1929
#define ROWS (Bb*Tt*Nn)       // 9936 tokens
#define GEMM_TM 32
#define GEMM_TN 64

typedef __attribute__((ext_vector_type(16))) _Float16 v16h;
typedef __attribute__((ext_vector_type(4)))  _Float16 v4h;
typedef __attribute__((ext_vector_type(4)))  float    v4f;
typedef __attribute__((ext_vector_type(8)))  float    v8f;

// ---------------------------------------------------------------------------
// Generic f32 GEMM through v_wmma_f32_16x16x32_f16.
// C[M x Nc] = act( A[M x K](lda) @ W[K x Nc](ldw) + bias )
// Block: 128 threads = 4 waves; tile 32(M) x 64(N); K stepped by 32.
// Full interior tiles take a guard-free vectorized path (float4 loads,
// packed b64 LDS stores in WMMA fragment order); edge tiles use clamped
// unconditional loads with a bitwise mask (not a select) so the compiler
// cannot sink loads under divergent branches.
// act: 0 = none, 1 = softplus
// ---------------------------------------------------------------------------
__global__ void k_gemm_wmma(const float* __restrict__ A, int lda,
                            const float* __restrict__ W, int ldw,
                            const float* __restrict__ bias, int act,
                            float* __restrict__ C, int ldc,
                            int M, int Nc, int K)
{
    // fragment-order tiles: Af[half][m(32)][j(16)], Bf[half][n(64)][j(16)]
    __shared__ __align__(32) _Float16 Af[2 * GEMM_TM * 16];
    __shared__ __align__(32) _Float16 Bf[2 * GEMM_TN * 16];

    const int m0   = blockIdx.y * GEMM_TM;
    const int n0   = blockIdx.x * GEMM_TN;
    const int tid  = threadIdx.x;     // 0..127
    const int wave = tid >> 5;        // 0..3  -> n sub-tile
    const int lane = tid & 31;
    const int half = lane >> 4;       // 0/1
    const int l16  = lane & 15;

    const int Mi = M - 1, Ki = K - 1, Ni = Nc - 1;
    const bool mn_full = (m0 + GEMM_TM <= M) && (n0 + GEMM_TN <= Nc) &&
                         ((lda & 3) == 0) && ((ldw & 3) == 0);

    v8f acc0 = {}, acc1 = {};

    for (int k0 = 0; k0 < K; k0 += 32) {
        if (mn_full && (k0 + 32 <= K)) {
            // ---------- fast path: no guards, vector loads ----------
            {   // A: thread handles row r, k-group kq*4 (2 passes of 16 rows)
                int kq  = tid & 7;
                int kr0 = kq * 4;
                int hf  = (kr0 >> 3) & 1;
                int k2  = kr0 - 8 * hf;
                int j0  = (k2 < 8) ? k2 : (k2 - 8);
#pragma unroll
                for (int p = 0; p < 2; ++p) {
                    int r = (tid >> 3) + p * 16;
                    v4f av = *reinterpret_cast<const v4f*>(
                        &A[(size_t)(m0 + r) * lda + k0 + kr0]);
                    v4h hp = { (_Float16)av.x, (_Float16)av.y,
                               (_Float16)av.z, (_Float16)av.w };
                    *reinterpret_cast<v4h*>(&Af[(hf * GEMM_TM + r) * 16 + j0]) = hp;
                }
            }
            {   // B: thread handles 4k x 4n sub-block
                int kq = tid >> 4;          // 0..7 -> kr = kq*4..+3
                int cq = tid & 15;          // 0..15 -> c = cq*4..+3
                int hf = (kq >> 1) & 1;
                int j0 = (kq & 1) * 4 + (kq >> 2) * 8;
                v4f rv[4];
#pragma unroll
                for (int dk = 0; dk < 4; ++dk)
                    rv[dk] = *reinterpret_cast<const v4f*>(
                        &W[(size_t)(k0 + kq * 4 + dk) * ldw + n0 + cq * 4]);
#pragma unroll
                for (int dc = 0; dc < 4; ++dc) {
                    float e0 = dc == 0 ? rv[0].x : dc == 1 ? rv[0].y : dc == 2 ? rv[0].z : rv[0].w;
                    float e1 = dc == 0 ? rv[1].x : dc == 1 ? rv[1].y : dc == 2 ? rv[1].z : rv[1].w;
                    float e2 = dc == 0 ? rv[2].x : dc == 1 ? rv[2].y : dc == 2 ? rv[2].z : rv[2].w;
                    float e3 = dc == 0 ? rv[3].x : dc == 1 ? rv[3].y : dc == 2 ? rv[3].z : rv[3].w;
                    v4h hp = { (_Float16)e0, (_Float16)e1, (_Float16)e2, (_Float16)e3 };
                    int c = cq * 4 + dc;
                    *reinterpret_cast<v4h*>(&Bf[(hf * GEMM_TN + c) * 16 + j0]) = hp;
                }
            }
            // prefetch next full K tile (working set is L2-resident)
            if (k0 + 64 <= K) {
                __builtin_prefetch(&A[(size_t)(m0 + (tid >> 3)) * lda +
                                      k0 + 32 + (tid & 7) * 4], 0, 3);
                __builtin_prefetch(&W[(size_t)(k0 + 32 + (tid >> 4) * 4) * ldw +
                                      n0 + (tid & 15) * 4], 0, 3);
            }
        } else {
            // ---------- edge path: clamped loads + bitwise mask ----------
#pragma unroll
            for (int it = 0; it < 8; ++it) {        // A: 1024 elems
                int idx = tid + it * 128;
                int r = idx >> 5, kr = idx & 31;
                int m = m0 + r, k = k0 + kr;
                int mc = m > Mi ? Mi : m;
                int kc = k > Ki ? Ki : k;
                int ok = (m <= Mi) & (k <= Ki);
                unsigned bits = __float_as_uint(A[(size_t)mc * lda + kc]) &
                                (unsigned)(-ok);
                int hf = (kr >> 3) & 1;
                int k2 = kr - 8 * hf;
                int j  = (k2 < 8) ? k2 : (k2 - 8);
                Af[(hf * GEMM_TM + r) * 16 + j] = (_Float16)__uint_as_float(bits);
            }
#pragma unroll
            for (int it = 0; it < 16; ++it) {       // B: 2048 elems
                int idx = tid + it * 128;
                int kr = idx >> 6, c = idx & 63;
                int k = k0 + kr, n = n0 + c;
                int kc = k > Ki ? Ki : k;
                int ncl = n > Ni ? Ni : n;
                int ok = (k <= Ki) & (n <= Ni);
                unsigned bits = __float_as_uint(W[(size_t)kc * ldw + ncl]) &
                                (unsigned)(-ok);
                int hf = (kr >> 3) & 1, j = (kr & 7) + ((kr >> 4) << 3);
                Bf[(hf * GEMM_TN + c) * 16 + j] = (_Float16)__uint_as_float(bits);
            }
        }
        __syncthreads();

        v16h a0 = *reinterpret_cast<const v16h*>(Af + (half * GEMM_TM + l16) * 16);
        v16h a1 = *reinterpret_cast<const v16h*>(Af + (half * GEMM_TM + 16 + l16) * 16);
        v16h b  = *reinterpret_cast<const v16h*>(
            Bf + (half * GEMM_TN + wave * 16 + l16) * 16);
        acc0 = __builtin_amdgcn_wmma_f32_16x16x32_f16(
            false, a0, false, b, (short)0, acc0, false, false);
        acc1 = __builtin_amdgcn_wmma_f32_16x16x32_f16(
            false, a1, false, b, (short)0, acc1, false, false);
        __syncthreads();
    }

    const int nc = n0 + wave * 16 + l16;
#pragma unroll
    for (int r = 0; r < 8; ++r) {
        int m = m0 + r + 8 * half;
        if (m < M && nc < Nc) {
            float v = acc0[r];
            if (bias) v += bias[nc];
            if (act == 1) v = (v > 20.0f) ? v : log1pf(expf(v));
            C[(size_t)m * ldc + nc] = v;
        }
        int m1 = m + 16;
        if (m1 < M && nc < Nc) {
            float v = acc1[r];
            if (bias) v += bias[nc];
            if (act == 1) v = (v > 20.0f) ? v : log1pf(expf(v));
            C[(size_t)m1 * ldc + nc] = v;
        }
    }
}

// ---------------------------------------------------------------------------
// Embedding: h = [x@W_in+b | tod_table[idx] | dow_table[idx] | adp_emb]
// ---------------------------------------------------------------------------
__global__ void k_embed(const float* __restrict__ x, const float* __restrict__ W_in,
                        const float* __restrict__ b_in, const float* __restrict__ tod_tab,
                        const float* __restrict__ dow_tab, const float* __restrict__ adp,
                        float* __restrict__ h)
{
    int idx = blockIdx.x * blockDim.x + threadIdx.x;
    if (idx >= ROWS * Dd) return;
    int d    = idx % Dd;
    int node = idx / Dd;          // bt*Nn + n
    int n    = node % Nn;
    int bt   = node / Nn;
    int t    = bt % Tt;
    const float* xr = x + (size_t)node * 3;
    float v;
    if (d < 24) {
        v = b_in[d] + xr[0] * W_in[d] + xr[1] * W_in[24 + d] + xr[2] * W_in[48 + d];
    } else if (d < 48) {
        int ti = (int)(xr[1] * (float)SPDc);
        ti = ti < 0 ? 0 : (ti > SPDc - 1 ? SPDc - 1 : ti);
        v = tod_tab[ti * 24 + (d - 24)];
    } else if (d < 72) {
        int di = (int)xr[2];
        di = di < 0 ? 0 : (di > 6 ? 6 : di);
        v = dow_tab[di * 24 + (d - 48)];
    } else {
        v = adp[((size_t)t * Nn + n) * 80 + (d - 72)];
    }
    h[idx] = v;
}

// one-shot init for GAT scratch: gout = 0, emax = -inf, den = 0
__global__ void k_gat_init(float* __restrict__ gout, float* __restrict__ emax,
                           float* __restrict__ den)
{
    for (int i = blockIdx.x * blockDim.x + threadIdx.x; i < ROWS * Dd;
         i += gridDim.x * blockDim.x) {
        gout[i] = 0.0f;
        if (i < Bb * Tt * Nn * Hh) { emax[i] = -INFINITY; den[i] = 0.0f; }
    }
}

// ---------------------------------------------------------------------------
// GAT pieces
// ---------------------------------------------------------------------------
__global__ void k_gat_scores(const float* __restrict__ xp,
                             const float* __restrict__ asrc,
                             const float* __restrict__ adst,
                             float* __restrict__ a_s, float* __restrict__ a_d)
{
    int idx = blockIdx.x * blockDim.x + threadIdx.x;   // (g*Nn+n)*Hh + h
    if (idx >= Bb * Tt * Nn * Hh) return;
    int hh = idx % Hh;
    int gn = idx / Hh;
    const float* row = xp + (size_t)gn * (Hh * Cc) + hh * Cc;
    float s0 = 0.f, s1 = 0.f;
    for (int c = 0; c < Cc; ++c) {
        float v = row[c];
        s0 += v * asrc[hh * Cc + c];
        s1 += v * adst[hh * Cc + c];
    }
    a_s[idx] = s0;
    a_d[idx] = s1;
}

__device__ inline void atomicMaxF(float* addr, float val)
{
    if (val >= 0.f) atomicMax((int*)addr, __float_as_int(val));
    else            atomicMin((unsigned int*)addr, __float_as_uint(val));
}

__device__ inline void edge_ends(const int* __restrict__ ei, int e, int& s, int& d)
{
    if (e < Ee) { s = ei[e]; d = ei[Ee + e]; }
    else        { s = e - Ee; d = e - Ee; }
}

__global__ void k_gat_emax(const int* __restrict__ ei, const float* __restrict__ a_s,
                           const float* __restrict__ a_d, float* __restrict__ emax)
{
    int idx = blockIdx.x * blockDim.x + threadIdx.x;   // (g*ETOT+e)*Hh + h
    if (idx >= Bb * Tt * ETOT * Hh) return;
    int hh = idx % Hh;
    int ge = idx / Hh;
    int e  = ge % ETOT;
    int g  = ge / ETOT;
    int s, d; edge_ends(ei, e, s, d);
    float v = a_s[(g * Nn + s) * Hh + hh] + a_d[(g * Nn + d) * Hh + hh];
    v = v > 0.f ? v : 0.2f * v;                        // leaky_relu
    atomicMaxF(&emax[(g * Nn + d) * Hh + hh], v);
}

__global__ void k_gat_expsum(const int* __restrict__ ei, const float* __restrict__ a_s,
                             const float* __restrict__ a_d, const float* __restrict__ emax,
                             float* __restrict__ ev, float* __restrict__ den)
{
    int idx = blockIdx.x * blockDim.x + threadIdx.x;
    if (idx >= Bb * Tt * ETOT * Hh) return;
    int hh = idx % Hh;
    int ge = idx / Hh;
    int e  = ge % ETOT;
    int g  = ge / ETOT;
    int s, d; edge_ends(ei, e, s, d);
    float v = a_s[(g * Nn + s) * Hh + hh] + a_d[(g * Nn + d) * Hh + hh];
    v = v > 0.f ? v : 0.2f * v;
    float ex = expf(v - emax[(g * Nn + d) * Hh + hh]);
    ev[idx] = ex;
    atomicAdd(&den[(g * Nn + d) * Hh + hh], ex);
}

__global__ void k_gat_agg(const int* __restrict__ ei, const float* __restrict__ xp,
                          const float* __restrict__ ev, const float* __restrict__ den,
                          float* __restrict__ gout)
{
    for (int idx = blockIdx.x * blockDim.x + threadIdx.x;
         idx < Bb * Tt * ETOT * Cc; idx += gridDim.x * blockDim.x) {
        int c  = idx % Cc;
        int ge = idx / Cc;
        int e  = ge % ETOT;
        int g  = ge / ETOT;
        int s, d; edge_ends(ei, e, s, d);
        float acc = 0.f;
#pragma unroll
        for (int hh = 0; hh < Hh; ++hh) {
            float alpha = ev[ge * Hh + hh] /
                          (den[(g * Nn + d) * Hh + hh] + 1e-16f);
            acc += alpha * xp[(size_t)(g * Nn + s) * (Hh * Cc) + hh * Cc + c];
        }
        atomicAdd(&gout[(size_t)(g * Nn + d) * Cc + c], acc * (1.0f / Hh));
    }
}

// ---------------------------------------------------------------------------
// X[row] = LayerNorm(X[row] + add[row] (+ bias)) * gamma + beta   (D = 152)
// ---------------------------------------------------------------------------
__global__ void k_add_ln(float* __restrict__ X, const float* __restrict__ add,
                         const float* __restrict__ bias,
                         const float* __restrict__ gam, const float* __restrict__ bet)
{
    __shared__ float s1[256];
    __shared__ float s2[256];
    int row = blockIdx.x;
    int d   = threadIdx.x;
    float v = 0.f;
    if (d < Dd) {
        v = X[(size_t)row * Dd + d] + add[(size_t)row * Dd + d];
        if (bias) v += bias[d];
    }
    s1[d] = v;
    s2[d] = v * v;
    __syncthreads();
    for (int off = 128; off > 0; off >>= 1) {
        if (d < off) { s1[d] += s1[d + off]; s2[d] += s2[d + off]; }
        __syncthreads();
    }
    float m   = s1[0] * (1.0f / Dd);
    float var = s2[0] * (1.0f / Dd) - m * m;
    float r   = rsqrtf(var + 1e-5f);
    if (d < Dd) X[(size_t)row * Dd + d] = (v - m) * r * gam[d] + bet[d];
}

// ---------------------------------------------------------------------------
// Mamba: depthwise causal conv(k=4, pad 3 left) + SiLU
// ---------------------------------------------------------------------------
__global__ void k_conv(const float* __restrict__ xz, const float* __restrict__ cw,
                       const float* __restrict__ cb, float* __restrict__ u,
                       int G, int L)
{
    int idx = blockIdx.x * blockDim.x + threadIdx.x;
    if (idx >= G * L * DINc) return;
    int c  = idx % DINc;
    int gl = idx / DINc;
    int l  = gl % L;
    int g  = gl / L;
    float acc = cb[c];
#pragma unroll
    for (int k = 0; k < 4; ++k) {
        int ls = l - 3 + k;
        if (ls >= 0)
            acc += xz[(size_t)(g * L + ls) * (2 * DINc) + c] * cw[c * 4 + k];
    }
    float sig = 1.f / (1.f + expf(-acc));
    u[(size_t)gl * DINc + c] = acc * sig;               // silu
}

// ---------------------------------------------------------------------------
// Selective scan, fused with D-skip and SiLU(z) gate
// ---------------------------------------------------------------------------
__global__ void k_scan(const float* __restrict__ u, const float* __restrict__ dt,
                       const float* __restrict__ dbl, const float* __restrict__ xz,
                       const float* __restrict__ Alog, const float* __restrict__ Dp,
                       float* __restrict__ prod, int G, int L)
{
    int idx = blockIdx.x * blockDim.x + threadIdx.x;
    if (idx >= G * DINc) return;
    int d = idx % DINc;
    int g = idx / DINc;
    float Arow[Sc];
#pragma unroll
    for (int s = 0; s < Sc; ++s) Arow[s] = -expf(Alog[d * Sc + s]);
    float Dv = Dp[d];
    float st[Sc];
#pragma unroll
    for (int s = 0; s < Sc; ++s) st[s] = 0.f;

    for (int l = 0; l < L; ++l) {
        size_t row = (size_t)(g * L + l);
        float dtv = dt[row * DINc + d];
        float uv  = u [row * DINc + d];
        const float* bl = dbl + row * (Rc + 2 * Sc);
        float y = 0.f;
#pragma unroll
        for (int s = 0; s < Sc; ++s) {
            float Bm = bl[Rc + s];
            float Cm = bl[Rc + Sc + s];
            st[s] = expf(dtv * Arow[s]) * st[s] + dtv * uv * Bm;
            y += st[s] * Cm;
        }
        float z   = xz[row * (2 * DINc) + DINc + d];
        float sig = 1.f / (1.f + expf(-z));
        prod[row * DINc + d] = (y + uv * Dv) * (z * sig);
    }
}

// ---------------------------------------------------------------------------
// Transposes (B,T,N,D) <-> (B,N,T,D), and final output permute
// ---------------------------------------------------------------------------
__global__ void k_t_fwd(const float* __restrict__ h, float* __restrict__ hT)
{
    for (int idx = blockIdx.x * blockDim.x + threadIdx.x; idx < ROWS * Dd;
         idx += gridDim.x * blockDim.x) {
        int d  = idx % Dd;
        int r  = idx / Dd;
        int n  = r % Nn;
        int bt = r / Nn;
        int t  = bt % Tt;
        int b  = bt / Tt;
        hT[((size_t)(b * Nn + n) * Tt + t) * Dd + d] = h[idx];
    }
}

__global__ void k_t_bwd(const float* __restrict__ hT, float* __restrict__ h)
{
    for (int idx = blockIdx.x * blockDim.x + threadIdx.x; idx < ROWS * Dd;
         idx += gridDim.x * blockDim.x) {
        int d  = idx % Dd;
        int r  = idx / Dd;
        int n  = r % Nn;
        int bt = r / Nn;
        int t  = bt % Tt;
        int b  = bt / Tt;
        h[idx] = hT[((size_t)(b * Nn + n) * Tt + t) * Dd + d];
    }
}

__global__ void k_outp(const float* __restrict__ tmp, float* __restrict__ out)
{
    int idx = blockIdx.x * blockDim.x + threadIdx.x;   // (b*Nn+n)*Tt + t
    if (idx >= Bb * Nn * Tt) return;
    int t  = idx % Tt;
    int bn = idx / Tt;
    int n  = bn % Nn;
    int b  = bn / Nn;
    out[(size_t)(b * Tt + t) * Nn + n] = tmp[idx];
}

// ---------------------------------------------------------------------------
// Host orchestration
// ---------------------------------------------------------------------------
extern "C" void kernel_launch(void* const* d_in, const int* in_sizes, int n_in,
                              void* d_out, int out_size, void* d_ws, size_t ws_size,
                              hipStream_t stream)
{
    (void)in_sizes; (void)n_in; (void)out_size; (void)ws_size;

    const float* x        = (const float*)d_in[0];
    const int*   ei       = (const int*)  d_in[1];
    const float* W_in     = (const float*)d_in[2];
    const float* b_in     = (const float*)d_in[3];
    const float* tod_tab  = (const float*)d_in[4];
    const float* dow_tab  = (const float*)d_in[5];
    const float* adp      = (const float*)d_in[6];
    const float* gat_W    = (const float*)d_in[7];
    const float* gat_asrc = (const float*)d_in[8];
    const float* gat_adst = (const float*)d_in[9];
    const float* gat_b    = (const float*)d_in[10];
    const float* gat_ln_g = (const float*)d_in[11];
    const float* gat_ln_b = (const float*)d_in[12];
    const float* mb_in_W  = (const float*)d_in[13];
    const float* mb_cw    = (const float*)d_in[14];
    const float* mb_cb    = (const float*)d_in[15];
    const float* mb_x_W   = (const float*)d_in[16];
    const float* mb_dt_W  = (const float*)d_in[17];
    const float* mb_dt_b  = (const float*)d_in[18];
    const float* mb_Alog  = (const float*)d_in[19];
    const float* mb_D     = (const float*)d_in[20];
    const float* mb_out_W = (const float*)d_in[21];
    const float* mb_ln_g  = (const float*)d_in[22];
    const float* mb_ln_b  = (const float*)d_in[23];
    const float* W_out    = (const float*)d_in[24];
    const float* b_out    = (const float*)d_in[25];

    // workspace layout (floats)
    float* ws = (float*)d_ws;
    const size_t SZ_H   = (size_t)ROWS * Dd;          // 1,510,272
    const size_t SZ_BIG = (size_t)ROWS * 2 * DINc;    // 6,041,088
    const size_t SZ_U   = (size_t)ROWS * DINc;        // 3,020,544
    const size_t SZ_DBL = (size_t)ROWS * (Rc + 2*Sc); //   417,312
    const size_t SZ_NH  = (size_t)Bb * Tt * Nn * Hh;  //    39,744
    const size_t SZ_EV  = (size_t)Bb * Tt * ETOT * Hh;//   370,368

    float* h    = ws;               float* hT  = h    + SZ_H;
    float* big  = hT  + SZ_H;       float* u   = big  + SZ_BIG;
    float* dbl  = u   + SZ_U;       float* dt  = dbl  + SZ_DBL;
    float* prod = dt  + SZ_U;       float* mout= prod + SZ_U;   // also GAT aggregate
    float* a_s  = mout+ SZ_H;       float* a_d = a_s  + SZ_NH;
    float* emax = a_d + SZ_NH;      float* den = emax + SZ_NH;
    float* ev   = den + SZ_NH;      // SZ_EV

    auto cdiv = [](int a, int b) { return (a + b - 1) / b; };
    auto gemm = [&](const float* A, int lda, const float* W, int ldw,
                    const float* bias, int act, float* Cp, int ldc,
                    int M, int Nc, int K) {
        dim3 g(cdiv(Nc, GEMM_TN), cdiv(M, GEMM_TM));
        k_gemm_wmma<<<g, 128, 0, stream>>>(A, lda, W, ldw, bias, act, Cp, ldc, M, Nc, K);
    };

    // 1) embedding
    k_embed<<<cdiv(ROWS * Dd, 256), 256, 0, stream>>>(x, W_in, b_in, tod_tab,
                                                      dow_tab, adp, h);

    for (int i = 0; i < 6; ++i) {
        // ---- GAT ----
        gemm(h, Dd, gat_W + (size_t)i * Dd * (Hh * Cc), Hh * Cc,
             nullptr, 0, big, Hh * Cc, ROWS, Hh * Cc, Dd);
        k_gat_scores<<<cdiv((int)SZ_NH, 256), 256, 0, stream>>>(
            big, gat_asrc + (size_t)i * Hh * Cc, gat_adst + (size_t)i * Hh * Cc, a_s, a_d);
        k_gat_init<<<cdiv((int)SZ_H, 256), 256, 0, stream>>>(mout, emax, den);
        k_gat_emax<<<cdiv((int)SZ_EV, 256), 256, 0, stream>>>(ei, a_s, a_d, emax);
        k_gat_expsum<<<cdiv((int)SZ_EV, 256), 256, 0, stream>>>(ei, a_s, a_d, emax, ev, den);
        k_gat_agg<<<cdiv(Bb * Tt * ETOT * Cc, 256), 256, 0, stream>>>(ei, big, ev, den, mout);
        k_add_ln<<<ROWS, 256, 0, stream>>>(h, mout, gat_b + (size_t)i * Cc,
                                           gat_ln_g + (size_t)i * Dd,
                                           gat_ln_b + (size_t)i * Dd);

        // ---- Mamba ----
        float* X = h; int G = Bb * Nn, L = Tt;        // flat view (828,12,152)
        if (i >= 3) {
            k_t_fwd<<<cdiv((int)SZ_H, 256), 256, 0, stream>>>(h, hT);
            X = hT; G = Bb * Tt; L = Nn;              // flat view (48,207,152)
        }
        gemm(X, Dd, mb_in_W + (size_t)i * Dd * 2 * DINc, 2 * DINc,
             nullptr, 0, big, 2 * DINc, ROWS, 2 * DINc, Dd);
        k_conv<<<cdiv((int)SZ_U, 256), 256, 0, stream>>>(
            big, mb_cw + (size_t)i * DINc * 4, mb_cb + (size_t)i * DINc, u, G, L);
        gemm(u, DINc, mb_x_W + (size_t)i * DINc * (Rc + 2 * Sc), Rc + 2 * Sc,
             nullptr, 0, dbl, Rc + 2 * Sc, ROWS, Rc + 2 * Sc, DINc);
        gemm(dbl, Rc + 2 * Sc, mb_dt_W + (size_t)i * Rc * DINc, DINc,
             mb_dt_b + (size_t)i * DINc, /*softplus*/1, dt, DINc, ROWS, DINc, Rc);
        k_scan<<<cdiv(G * DINc, 256), 256, 0, stream>>>(
            u, dt, dbl, big, mb_Alog + (size_t)i * DINc * Sc,
            mb_D + (size_t)i * DINc, prod, G, L);
        gemm(prod, DINc, mb_out_W + (size_t)i * DINc * Dd, Dd,
             nullptr, 0, mout, Dd, ROWS, Dd, DINc);
        k_add_ln<<<ROWS, 256, 0, stream>>>(X, mout, nullptr,
                                           mb_ln_g + (size_t)i * Dd,
                                           mb_ln_b + (size_t)i * Dd);
        if (i >= 3)
            k_t_bwd<<<cdiv((int)SZ_H, 256), 256, 0, stream>>>(hT, h);
    }

    // 3) output head: (B,N,T*D) @ W_out + b_out -> permute to (B,T,N,1)
    k_t_fwd<<<cdiv((int)SZ_H, 256), 256, 0, stream>>>(h, hT);
    gemm(hT, Tt * Dd, W_out, Tt, b_out, 0, dbl /*tmp 828x12*/, Tt,
         Bb * Nn, Tt, Tt * Dd);
    k_outp<<<cdiv(Bb * Nn * Tt, 256), 256, 0, stream>>>(dbl, (float*)d_out);
}
